// MultiHeadAttention_78331613544747
// MI455X (gfx1250) — compile-verified, matrix operand order checked by reading, not run on hardware
//
#include <hip/hip_runtime.h>
#include <hip/hip_bf16.h>

#define B_ 8
#define S_ 1024
#define D_ 768
#define H_ 12
#define DH_ 64

typedef __attribute__((ext_vector_type(16))) __bf16 v16bf;
typedef __attribute__((ext_vector_type(8)))  __bf16 v8bf;
typedef __attribute__((ext_vector_type(8)))  float  v8f;

__device__ __forceinline__ v8f wmma_bf16(v16bf a, v16bf b, v8f c) {
  // D = A(16x32 bf16) * B(32x16 bf16) + C(16x16 f32)
  return __builtin_amdgcn_wmma_f32_16x16x32_bf16(false, a, false, b, (short)0, c,
                                                 false, false);
}

// Load one 16x32 bf16 fragment. Convention (per ISA 16-bit A layout, and the
// symmetric B layout): "row" r = lane&15 (M for A-frags, N for B-frags),
// contiguous memory dim = K. Lanes 0-15 hold K={0..7,16..23}, lanes 16-31 hold
// K={8..15,24..31}. Two contiguous 16B loads per lane.
__device__ __forceinline__ v16bf load_frag(const __bf16* base, int rstride) {
  const int lane = threadIdx.x & 31;
  const __bf16* p = base + (lane & 15) * rstride + ((lane >> 4) << 3);
  v8bf lo = *(const v8bf*)(p);
  v8bf hi = *(const v8bf*)(p + 16);
  v16bf f;
#pragma unroll
  for (int i = 0; i < 8; ++i) { f[i] = lo[i]; f[i + 8] = hi[i]; }
  return f;
}

// ---------------- conversion kernels ----------------
__global__ void cvt_f32_bf16(const float* __restrict__ src,
                             __bf16* __restrict__ dst, int n) {
  int i = blockIdx.x * blockDim.x + threadIdx.x;
  int stride = gridDim.x * blockDim.x;
  for (; i < n; i += stride) dst[i] = (__bf16)src[i];
}

// [H][D][DH] f32 -> [H][DH][D] bf16 (transpose last two dims)
__global__ void cvt_w_transpose(const float* __restrict__ src,
                                __bf16* __restrict__ dst) {
  int i = blockIdx.x * blockDim.x + threadIdx.x;
  const int n = H_ * D_ * DH_;
  int stride = gridDim.x * blockDim.x;
  for (; i < n; i += stride) {
    int h = i / (D_ * DH_);
    int r = i - h * (D_ * DH_);
    int d = r / DH_;
    int e = r - d * DH_;
    dst[(h * DH_ + e) * D_ + d] = (__bf16)src[i];
  }
}

// ---------------- QKV projection: [S,D] @ [D,DH] per (b,h,mat) ----------------
__global__ __launch_bounds__(256) void qkv_proj(
    const __bf16* __restrict__ xb, const __bf16* __restrict__ wqT,
    const __bf16* __restrict__ wkT, const __bf16* __restrict__ wvT,
    const float* __restrict__ bq, const float* __restrict__ bk,
    const float* __restrict__ bv, __bf16* __restrict__ qb,
    __bf16* __restrict__ kb, __bf16* __restrict__ vT) {
  const int b = blockIdx.z;
  const int h = blockIdx.y % H_;
  const int mat = blockIdx.y / H_;          // 0=q 1=k 2=v
  const int rowbase = blockIdx.x * 64;
  const int wave = threadIdx.x >> 5;
  const int lane = threadIdx.x & 31;
  const int ms = wave & 3;                  // 16-row subtile
  const int nsb = (wave >> 2) << 1;         // n-subtile pair {0,1} or {2,3}
  const __bf16* wT = (mat == 0) ? wqT : (mat == 1) ? wkT : wvT;
  const float* bias = (mat == 0) ? bq : (mat == 1) ? bk : bv;
  const __bf16* arow = xb + (size_t)(b * S_ + rowbase + ms * 16) * D_;
  const __bf16* w0 = wT + (size_t)(h * DH_ + nsb * 16) * D_;
  const __bf16* w1 = w0 + (size_t)16 * D_;
  v8f acc0 = {}; v8f acc1 = {};
  for (int k = 0; k < D_; k += 32) {
    __builtin_prefetch(arow + k + 32, 0, 1);
    v16bf a = load_frag(arow + k, D_);
    acc0 = wmma_bf16(a, load_frag(w0 + k, D_), acc0);
    acc1 = wmma_bf16(a, load_frag(w1 + k, D_), acc1);
  }
  const float scale = (mat == 0) ? 0.125f : 1.0f;  // q pre-scaled by 1/sqrt(64)
  const int mofs = (lane >> 4) << 3;
  for (int t = 0; t < 2; ++t) {
    v8f acc = t ? acc1 : acc0;
    const int e = (nsb + t) * 16 + (lane & 15);
    const float bi = bias[h * DH_ + e];
    if (mat < 2) {
      __bf16* dst = (mat == 0 ? qb : kb) +
          (size_t)((b * H_ + h) * S_ + rowbase + ms * 16 + mofs) * DH_ + e;
#pragma unroll
      for (int j = 0; j < 8; ++j)
        dst[(size_t)j * DH_] = (__bf16)((acc[j] + bi) * scale);
    } else {
      v8bf pk;
#pragma unroll
      for (int j = 0; j < 8; ++j) pk[j] = (__bf16)(acc[j] + bi);
      // transposed store: vT[b][h][e][s], 8 consecutive s -> one 16B store
      __bf16* dst = vT + (size_t)((b * H_ + h) * DH_ + e) * S_ +
                    rowbase + ms * 16 + mofs;
      *(v8bf*)dst = pk;
    }
  }
}

// ---------------- attention: scores -> softmax -> P@V per (b,h,32-row tile) ----
__global__ __launch_bounds__(256) void attn_kernel(
    const __bf16* __restrict__ qb, const __bf16* __restrict__ kb,
    const __bf16* __restrict__ vT, __bf16* __restrict__ ob) {
  extern __shared__ __align__(16) char smem_raw[];
  float*  sc = (float*)smem_raw;                         // [32][1024] f32
  __bf16* pb = (__bf16*)(smem_raw + 32 * 1024 * 4);      // [32][1024] bf16
  const int b = blockIdx.z, h = blockIdx.y;
  const int rowbase = blockIdx.x * 32;
  const int wave = threadIdx.x >> 5;
  const int lane = threadIdx.x & 31;
  const size_t bh = (size_t)(b * H_ + h);

  // ---- scores = q @ k^T (scale already folded into q) ----
  {
    const int strip = wave & 1;    // which 16-row strip
    const int cg = wave >> 1;      // col group: 16 col-tiles each
    const __bf16* qrow = qb + (bh * S_ + rowbase + strip * 16) * DH_;
    v16bf a0 = load_frag(qrow, DH_);        // K = 0..31
    v16bf a1 = load_frag(qrow + 32, DH_);   // K = 32..63
    for (int ct = cg * 16; ct < cg * 16 + 16; ++ct) {
      const __bf16* krow = kb + (bh * S_ + ct * 16) * DH_;
      v8f acc = {};
      acc = wmma_bf16(a0, load_frag(krow, DH_), acc);
      acc = wmma_bf16(a1, load_frag(krow + 32, DH_), acc);
      const int n = lane & 15, mofs = (lane >> 4) << 3;
      float* srow = sc + (strip * 16 + mofs) * 1024 + ct * 16 + n;
#pragma unroll
      for (int j = 0; j < 8; ++j) srow[j * 1024] = acc[j];
    }
  }
  __syncthreads();

  // ---- row softmax: 8 lanes per row, 128 elements each ----
  {
    const int r = threadIdx.x >> 3;
    const int seg = threadIdx.x & 7;
    float* row = sc + r * 1024 + seg * 128;
    float mx = -3.402823466e38f;
#pragma unroll 4
    for (int i = 0; i < 128; ++i) mx = fmaxf(mx, row[i]);
#pragma unroll
    for (int o = 1; o < 8; o <<= 1) mx = fmaxf(mx, __shfl_xor(mx, o, 8));
    float sum = 0.f;
#pragma unroll 4
    for (int i = 0; i < 128; ++i) {
      float e = __expf(row[i] - mx);
      row[i] = e;
      sum += e;
    }
#pragma unroll
    for (int o = 1; o < 8; o <<= 1) sum += __shfl_xor(sum, o, 8);
    const float inv = 1.0f / sum;
    __bf16* prow = pb + r * 1024 + seg * 128;
#pragma unroll 4
    for (int i = 0; i < 128; ++i) prow[i] = (__bf16)(row[i] * inv);
  }
  __syncthreads();

  // ---- out = P @ V : each wave owns one 16x16 output subtile, K=1024 ----
  {
    const int ms = wave & 1;
    const int ns = wave >> 1;    // 0..3
    v8f acc = {};
    const __bf16* pbase = pb + (ms * 16) * 1024;
    const __bf16* vbase = vT + (bh * DH_ + ns * 16) * S_;
    for (int kk = 0; kk < 1024; kk += 32) {
      v16bf a = load_frag(pbase + kk, 1024);   // LDS: ds_load_b128 x2
      v16bf bb = load_frag(vbase + kk, S_);    // global, K-contiguous
      acc = wmma_bf16(a, bb, acc);
    }
    const int e = ns * 16 + (lane & 15);
    const int s0 = rowbase + ms * 16 + ((lane >> 4) << 3);
#pragma unroll
    for (int j = 0; j < 8; ++j)
      ob[((size_t)(b * S_ + s0 + j)) * D_ + h * DH_ + e] = (__bf16)acc[j];
  }
}

// ---------------- output projection: [B*S, D] @ Wo^T + bo -> f32 ----------------
__global__ __launch_bounds__(256) void out_proj(
    const __bf16* __restrict__ ab, const __bf16* __restrict__ wo,
    const float* __restrict__ bo, float* __restrict__ out) {
  const int rowbase = blockIdx.x * 64;
  const int colbase = blockIdx.y * 64;
  const int wave = threadIdx.x >> 5;
  const int lane = threadIdx.x & 31;
  const int ms = wave & 3;
  const int nsb = (wave >> 2) << 1;
  const __bf16* arow = ab + (size_t)(rowbase + ms * 16) * D_;
  // B(k=d, n=d') = Wo[d'][d] -> row-major Wo is already n-major, K-contiguous
  const __bf16* w0 = wo + (size_t)(colbase + nsb * 16) * D_;
  const __bf16* w1 = w0 + (size_t)16 * D_;
  v8f acc0 = {}, acc1 = {};
  for (int k = 0; k < D_; k += 32) {
    __builtin_prefetch(arow + k + 32, 0, 1);
    v16bf a = load_frag(arow + k, D_);
    acc0 = wmma_bf16(a, load_frag(w0 + k, D_), acc0);
    acc1 = wmma_bf16(a, load_frag(w1 + k, D_), acc1);
  }
  const int mofs = (lane >> 4) << 3;
  for (int t = 0; t < 2; ++t) {
    v8f acc = t ? acc1 : acc0;
    const int n = colbase + (nsb + t) * 16 + (lane & 15);
    const float bi = bo[n];
    float* dst = out + (size_t)(rowbase + ms * 16 + mofs) * D_ + n;
#pragma unroll
    for (int j = 0; j < 8; ++j) dst[(size_t)j * D_] = acc[j] + bi;
  }
}

extern "C" void kernel_launch(void* const* d_in, const int* in_sizes, int n_in,
                              void* d_out, int out_size, void* d_ws,
                              size_t ws_size, hipStream_t stream) {
  (void)in_sizes; (void)n_in; (void)out_size; (void)ws_size;
  const float* x  = (const float*)d_in[0];
  const float* Wq = (const float*)d_in[1];
  const float* bq = (const float*)d_in[2];
  const float* Wk = (const float*)d_in[3];
  const float* bk = (const float*)d_in[4];
  const float* Wv = (const float*)d_in[5];
  const float* bv = (const float*)d_in[6];
  const float* Wo = (const float*)d_in[7];
  const float* bo = (const float*)d_in[8];
  float* out = (float*)d_out;

  char* ws = (char*)d_ws;
  size_t off = 0;
  auto carve = [&](size_t bytes) -> void* {
    void* p = ws + off;
    off += (bytes + 255) & ~(size_t)255;
    return p;
  };
  __bf16* xb  = (__bf16*)carve((size_t)B_ * S_ * D_ * 2);
  __bf16* wqT = (__bf16*)carve((size_t)H_ * DH_ * D_ * 2);
  __bf16* wkT = (__bf16*)carve((size_t)H_ * DH_ * D_ * 2);
  __bf16* wvT = (__bf16*)carve((size_t)H_ * DH_ * D_ * 2);
  __bf16* wob = (__bf16*)carve((size_t)D_ * D_ * 2);
  __bf16* qb  = (__bf16*)carve((size_t)B_ * H_ * S_ * DH_ * 2);
  __bf16* kb  = (__bf16*)carve((size_t)B_ * H_ * S_ * DH_ * 2);
  __bf16* vT  = (__bf16*)carve((size_t)B_ * H_ * DH_ * S_ * 2);
  __bf16* ab  = (__bf16*)carve((size_t)B_ * S_ * D_ * 2);

  cvt_f32_bf16<<<2048, 256, 0, stream>>>(x, xb, B_ * S_ * D_);
  cvt_f32_bf16<<<512, 256, 0, stream>>>(Wo, wob, D_ * D_);
  cvt_w_transpose<<<1024, 256, 0, stream>>>(Wq, wqT);
  cvt_w_transpose<<<1024, 256, 0, stream>>>(Wk, wkT);
  cvt_w_transpose<<<1024, 256, 0, stream>>>(Wv, wvT);

  qkv_proj<<<dim3(S_ / 64, H_ * 3, B_), 256, 0, stream>>>(
      xb, wqT, wkT, wvT, bq, bk, bv, qb, kb, vT);
  attn_kernel<<<dim3(S_ / 32, H_, B_), 256, 32 * 1024 * 4 + 32 * 1024 * 2,
                stream>>>(qb, kb, vT, ab);
  out_proj<<<dim3((B_ * S_) / 64, D_ / 64, 1), 256, 0, stream>>>(ab, wob, bo,
                                                                 out);
}
